// FractalAttention_70617852281036
// MI455X (gfx1250) — matrix operand order repaired and checked by reading, not compiled, not run on hardware
//
#include <hip/hip_runtime.h>
#include <hip/hip_bf16.h>

// ---------------------------------------------------------------------------
// FractalAttention on MI455X (gfx1250, wave32, WMMA + async-LDS)
//   pre-pass   : fp32 -> bf16 for x and all four weight matrices (once)
//   projections: v_wmma_f32_16x16x32_bf16, double-buffered
//                global_load_async_to_lds_b128 staging (ASYNCcnt)
//   attention  : wave32 VALU + shuffle softmax (per-query gathered K/V rows
//                cannot share a WMMA B operand)
//   output proj: same WMMA GEMM consuming bf16 attention output
// Roofline: ~17.5 GFLOP GEMM vs ~85 MB of bf16 traffic -> bandwidth bound;
// bf16 halves GEMM reads, async-LDS removes VGPR round-trips in staging.
// ---------------------------------------------------------------------------

#define BB   2
#define SS   4096
#define DD   512
#define HH   8
#define DH   64
#define WW   32
#define MTOT (BB * SS)          // 8192 rows in every GEMM
#define LSTR 40                 // LDS row stride in ushorts: 80 B (16B-aligned)

typedef __bf16 bf16x16 __attribute__((ext_vector_type(16)));
typedef float  f32x8   __attribute__((ext_vector_type(8)));
typedef int    v4i_ld  __attribute__((vector_size(16)));   // builtin pointee type

union Frag16 {
    bf16x16  v;
    unsigned u[8];
};

__device__ __forceinline__ unsigned short f32_to_bf16(float f) {
    unsigned u = __builtin_bit_cast(unsigned, f);
    unsigned r = 0x7FFFu + ((u >> 16) & 1u);     // round-to-nearest-even
    return (unsigned short)((u + r) >> 16);
}

// --- gfx1250 async global->LDS path (guarded so the build never breaks) ----
#if defined(__gfx1250__) && __has_builtin(__builtin_amdgcn_global_load_async_to_lds_b128)
#define HAVE_ASYNC 1
#else
#define HAVE_ASYNC 0
#endif

#if HAVE_ASYNC
#if __has_builtin(__builtin_amdgcn_s_wait_asynccnt)
#define ASYNC_WAIT0() __builtin_amdgcn_s_wait_asynccnt(0)
#else
#define ASYNC_WAIT0() asm volatile("s_wait_asynccnt 0x0" ::: "memory")
#endif
// builtin signature (from hipcc diagnostic): (v4i AS1*, v4i AS3*, Ii off, Ii cpol)
#define GLD_ASYNC_B128(gsrc, ldst)                                             \
    __builtin_amdgcn_global_load_async_to_lds_b128(                            \
        (__attribute__((address_space(1))) v4i_ld*)(gsrc),                     \
        (__attribute__((address_space(3))) v4i_ld*)(ldst), 0, 0)
#else
#define ASYNC_WAIT0() ((void)0)
#endif

// ---------------------------------------------------------------------------
// f32 -> bf16 bulk convert, 4 elems/thread (float4 in, 8B packed out)
// ---------------------------------------------------------------------------
__global__ __launch_bounds__(256) void cvt_f32_bf16(
    const float* __restrict__ in, unsigned short* __restrict__ out, int n4)
{
    const int i = blockIdx.x * blockDim.x + threadIdx.x;
    if (i >= n4) return;
    float4 f = ((const float4*)in)[i];
    union { unsigned short s[4]; uint2 u; } p;
    p.s[0] = f32_to_bf16(f.x);
    p.s[1] = f32_to_bf16(f.y);
    p.s[2] = f32_to_bf16(f.z);
    p.s[3] = f32_to_bf16(f.w);
    *(uint2*)(out + (size_t)i * 4) = p.u;
}

// ---------------------------------------------------------------------------
// C(M x 512) = A(M x 512, bf16) * W^T (W 512x512 bf16, row-major out_dim x K)
// Block 256 thr = 8 waves; C tile 128(M) x 64(N); wave = 16 rows x 4 tiles.
// Double-buffered LDS, async B128 staging when available.
// ---------------------------------------------------------------------------
__global__ __launch_bounds__(256) void gemm512_bf16_wmma(
    const unsigned short* __restrict__ A,
    const unsigned short* __restrict__ Wm,
    float* __restrict__ C)
{
    __shared__ unsigned short lx[2][128 * LSTR];   // A tiles 128 x 32
    __shared__ unsigned short lw[2][64 * LSTR];    // B cols  64 x 32

    const int tid  = threadIdx.x;
    const int lane = tid & 31;
    const int wv   = tid >> 5;               // wave 0..7
    const int half = lane >> 4;              // ISA 16-bit fragment half-select
    const int r    = lane & 15;

    const int bm = blockIdx.y;               // 64 blocks of 128 rows
    const int bn = blockIdx.x;               // 8 blocks of 64 cols

    // 16B chunk assignments: A tile = 512 chunks (2/thread), W tile = 256 (1/thread)
    const int ac  = tid * 2;
    const int ar0 = ac >> 2,        ak0 = (ac & 3) * 8;
    const int ar1 = (ac + 1) >> 2,  ak1 = ((ac + 1) & 3) * 8;
    const int wr  = tid >> 2,       wk  = (tid & 3) * 8;

    const unsigned short* Abase = A  + (size_t)(bm * 128) * DD;
    const unsigned short* Wbase = Wm + (size_t)(bn * 64)  * DD;

    f32x8 acc[4] = {};

    auto issue = [&](int kk, int buf) {
#if HAVE_ASYNC
        GLD_ASYNC_B128(Abase + (size_t)ar0 * DD + kk + ak0,
                       &lx[buf][ar0 * LSTR + ak0]);
        GLD_ASYNC_B128(Abase + (size_t)ar1 * DD + kk + ak1,
                       &lx[buf][ar1 * LSTR + ak1]);
        GLD_ASYNC_B128(Wbase + (size_t)wr * DD + kk + wk,
                       &lw[buf][wr * LSTR + wk]);
#else
        {
            const unsigned* s = (const unsigned*)(Abase + (size_t)ar0 * DD + kk + ak0);
            unsigned short* d = &lx[buf][ar0 * LSTR + ak0];
#pragma unroll
            for (int j = 0; j < 4; ++j) *(unsigned*)(d + 2 * j) = s[j];
        }
        {
            const unsigned* s = (const unsigned*)(Abase + (size_t)ar1 * DD + kk + ak1);
            unsigned short* d = &lx[buf][ar1 * LSTR + ak1];
#pragma unroll
            for (int j = 0; j < 4; ++j) *(unsigned*)(d + 2 * j) = s[j];
        }
        {
            const unsigned* s = (const unsigned*)(Wbase + (size_t)wr * DD + kk + wk);
            unsigned short* d = &lw[buf][wr * LSTR + wk];
#pragma unroll
            for (int j = 0; j < 4; ++j) *(unsigned*)(d + 2 * j) = s[j];
        }
#endif
    };

    issue(0, 0);

    for (int step = 0; step < DD / 32; ++step) {
        const int buf = step & 1;
        ASYNC_WAIT0();               // current buffer's async loads complete
        __syncthreads();             // ...for every wave in the group
        if (step + 1 < DD / 32)
            issue((step + 1) * 32, buf ^ 1);   // overlap next tile with WMMA

        // ---- A fragment (16-bit A 16x32 ISA layout) ----
        Frag16 af;
        {
            const unsigned short* rowp = &lx[buf][(wv * 16 + r) * LSTR];
#pragma unroll
            for (int v = 0; v < 4; ++v)      // K = half*8 + {0..7}
                af.u[v] = *(const unsigned*)(rowp + half * 8 + 2 * v);
#pragma unroll
            for (int v = 0; v < 4; ++v)      // K = 16 + half*8 + {0..7}
                af.u[4 + v] = *(const unsigned*)(rowp + 16 + half * 8 + 2 * v);
        }

        // ---- 4 N-tiles: B fragment (32x16 col layout) + WMMA ----
#pragma unroll
        for (int t = 0; t < 4; ++t) {
            Frag16 bfr;
            const unsigned short* colp = &lw[buf][(t * 16 + r) * LSTR + half * 8];
#pragma unroll
            for (int v = 0; v < 8; ++v)
                bfr.u[v] = *(const unsigned*)(colp + 2 * v + (v & 4) * 2);

            acc[t] = __builtin_amdgcn_wmma_f32_16x16x32_bf16(
                false, af.v, false, bfr.v, (short)0, acc[t], false, false);
        }
        __syncthreads();             // reads done before buffer reuse
    }

    // ---- store C per ISA C/D layout: VGPR j -> row half*8 + j, col lane&15
#pragma unroll
    for (int t = 0; t < 4; ++t) {
        const int gcol = bn * 64 + t * 16 + r;
#pragma unroll
        for (int j = 0; j < 8; ++j) {
            const int grow = bm * 128 + wv * 16 + half * 8 + j;
            C[(size_t)grow * DD + gcol] = acc[t][j];
        }
    }
}

// ---------------------------------------------------------------------------
// Neighborhood attention: one wave32 per (b, s, h).
//   phase 1: lane = neighbor w  -> score dot(q, k_nbr) * 1/sqrt(64)
//   softmax: shuffle-reduce max & sum across the 32 lanes
//   phase 2: lane = output dims (d, d+32) -> sum_w p[w] * V[nbr[w]][d]
// Output written bf16 so the W_o GEMM consumes it via WMMA directly.
// ---------------------------------------------------------------------------
__global__ __launch_bounds__(256) void fractal_attn_wave(
    const float* __restrict__ Q, const float* __restrict__ K,
    const float* __restrict__ V, const int* __restrict__ nbr,
    unsigned short* __restrict__ AOut)
{
    const int wid  = (int)((blockIdx.x * blockDim.x + threadIdx.x) >> 5);
    const int lane = threadIdx.x & 31;

    const int h = wid & (HH - 1);
    const int s = (wid >> 3) & (SS - 1);
    const int b = wid >> 15;                       // S*H = 2^15

    const size_t qoff = ((size_t)(b * SS + s)) * DD + h * DH;

    const int nb = nbr[s * WW + lane];             // this lane's neighbor
    const size_t koff = ((size_t)(b * SS + nb)) * DD + h * DH;

    float sc = 0.f;
#pragma unroll 8
    for (int d = 0; d < DH; ++d)
        sc = __builtin_fmaf(Q[qoff + d], K[koff + d], sc);
    sc *= 0.125f;                                  // 1/sqrt(64)

    float m = sc;
#pragma unroll
    for (int off = 16; off > 0; off >>= 1)
        m = fmaxf(m, __shfl_xor(m, off, 32));
    float p = __expf(sc - m);
    float sum = p;
#pragma unroll
    for (int off = 16; off > 0; off >>= 1)
        sum += __shfl_xor(sum, off, 32);
    p = p / sum;

    float a0 = 0.f, a1 = 0.f;
#pragma unroll 4
    for (int w = 0; w < WW; ++w) {
        const float aw = __shfl(p, w, 32);
        const int   iw = __shfl(nb, w, 32);
        const float* vr = V + ((size_t)(b * SS + iw)) * DD + h * DH;
        a0 = __builtin_fmaf(aw, vr[lane], a0);
        a1 = __builtin_fmaf(aw, vr[lane + 32], a1);
    }
    AOut[qoff + lane]      = f32_to_bf16(a0);
    AOut[qoff + lane + 32] = f32_to_bf16(a1);
}

// ---------------------------------------------------------------------------
extern "C" void kernel_launch(void* const* d_in, const int* in_sizes, int n_in,
                              void* d_out, int out_size, void* d_ws, size_t ws_size,
                              hipStream_t stream)
{
    const float* x   = (const float*)d_in[0];
    const float* Wq  = (const float*)d_in[1];
    const float* Wk  = (const float*)d_in[2];
    const float* Wv  = (const float*)d_in[3];
    const float* Wo  = (const float*)d_in[4];
    const int*   nbr = (const int*)d_in[5];
    float* out = (float*)d_out;

    // workspace layout:
    //  xh (8.4MB) | Wq/Wk/Wv/Wo bf16 (4 x 0.5MB) | Q,K,V fp32 (3 x 16.8MB)
    //  | attn out bf16 (8.4MB)   -> ~66 MB total
    unsigned short* xh  = (unsigned short*)d_ws;
    unsigned short* Wqh = xh  + (size_t)MTOT * DD;
    unsigned short* Wkh = Wqh + (size_t)DD * DD;
    unsigned short* Wvh = Wkh + (size_t)DD * DD;
    unsigned short* Woh = Wvh + (size_t)DD * DD;
    float* Q = (float*)(Woh + (size_t)DD * DD);
    float* K = Q + (size_t)MTOT * DD;
    float* V = K + (size_t)MTOT * DD;
    unsigned short* AO = (unsigned short*)(V + (size_t)MTOT * DD);

    const dim3 blk(256);

    // ---- one-time bf16 conversion (bandwidth-bound op: halve GEMM reads) --
    cvt_f32_bf16<<<dim3((MTOT * DD) / 4 / 256), blk, 0, stream>>>(x, xh, (MTOT * DD) / 4);
    cvt_f32_bf16<<<dim3((DD * DD) / 4 / 256), blk, 0, stream>>>(Wq, Wqh, (DD * DD) / 4);
    cvt_f32_bf16<<<dim3((DD * DD) / 4 / 256), blk, 0, stream>>>(Wk, Wkh, (DD * DD) / 4);
    cvt_f32_bf16<<<dim3((DD * DD) / 4 / 256), blk, 0, stream>>>(Wv, Wvh, (DD * DD) / 4);
    cvt_f32_bf16<<<dim3((DD * DD) / 4 / 256), blk, 0, stream>>>(Wo, Woh, (DD * DD) / 4);

    // ---- projections -----------------------------------------------------
    const dim3 grid(DD / 64, MTOT / 128);          // (8, 64)
    gemm512_bf16_wmma<<<grid, blk, 0, stream>>>(xh, Wqh, Q);
    gemm512_bf16_wmma<<<grid, blk, 0, stream>>>(xh, Wkh, K);
    gemm512_bf16_wmma<<<grid, blk, 0, stream>>>(xh, Wvh, V);

    // ---- neighborhood attention -----------------------------------------
    fractal_attn_wave<<<dim3((MTOT * HH) / 8), blk, 0, stream>>>(Q, K, V, nbr, AO);

    // ---- output projection ----------------------------------------------
    gemm512_bf16_wmma<<<grid, blk, 0, stream>>>(AO, Woh, out);
}
